// Decoder_67095979098399
// MI455X (gfx1250) — matrix operand order repaired and checked by reading, not compiled, not run on hardware
//
#include <hip/hip_runtime.h>
#include <hip/hip_bf16.h>
#include <math.h>

// ---------------------------------------------------------------------------
// LSTM greedy decoder for MI455X (gfx1250, wave32, WMMA bf16 path).
//   B=64, H=E=512, V=32000, T=64.
// Per step (3 kernels): fused gates-GEMM+cell (WMMA bf16, LDS gate tile),
// logits GEMM + fused per-block argmax partials (WMMA bf16, h staged in LDS,
// W_out bf16 L2-resident), argmax finalize + embedding gather.
// All GEMM K-loops are software-pipelined (prefetch next fragments before
// issuing WMMAs) so waits are partial instead of s_wait_loadcnt 0.
// ---------------------------------------------------------------------------

typedef __bf16 bf16;
typedef __attribute__((ext_vector_type(16))) __bf16 v16bf;
typedef __attribute__((ext_vector_type(8)))  __bf16 v8bf;
typedef __attribute__((ext_vector_type(8)))  float  v8f;

#define BATCH 64
#define HID   512
#define EMB   512
#define VOCAB 32000
#define TLEN  64
#define G4    (4 * HID)        // 2048
#define NLBLK (VOCAB / 128)    // 250 logits blocks

union Frag { v16bf v; v8bf h[2]; };

// A-fragment (16x32 bf16, M x K) from row-major [rows][ld] bf16 in global.
// lane<16 : row = row0+lane,    K-chunks k0+0..7  and k0+16..23
// lane>=16: row = row0+lane-16, K-chunks k0+8..15 and k0+24..31
__device__ inline v16bf load_frag_a(const bf16* __restrict__ base, int ld,
                                    int row0, int k0, int l, int hi) {
  const bf16* p = base + (size_t)(row0 + l) * ld + k0 + hi * 8;
  Frag f;
  f.h[0] = *(const v8bf*)(p);
  f.h[1] = *(const v8bf*)(p + 16);
  return f.v;
}

// B-fragment (32x16 bf16, K x N) with B[k][n] = W[n][k], W row-major [N][ld].
// lane<16 : col = n0+lane,    K-chunks k0+0..7  and k0+16..23
// lane>=16: col = n0+lane-16, K-chunks k0+8..15 and k0+24..31
__device__ inline v16bf load_frag_b(const bf16* __restrict__ w, int ld,
                                    int n0, int k0, int l, int hi) {
  const bf16* p = w + (size_t)(n0 + l) * ld + k0 + hi * 8;
  Frag f;
  f.h[0] = *(const v8bf*)(p);
  f.h[1] = *(const v8bf*)(p + 16);
  return f.v;
}

#define WMMA_BF16(A, B, C) \
  __builtin_amdgcn_wmma_f32_16x16x32_bf16(false, (A), false, (B), (short)0, (C), false, false)

// ---------------------------------------------------------------------------
// Fused gates GEMM + LSTM cell.
// Block = 32 j-columns x all 4 gates (8 waves: gate = wave>>1, coltile = wave&1).
// gates = x @ W_ih^T + h @ W_hh^T + b_ih + b_hh staged in 32KB LDS, then
// c' = sig(f)*c + sig(i)*tanh(g); h' = sig(o)*tanh(c') written in-block.
// Reads hin (h_{t-1}), writes hout (h_t) -> double-buffered by caller.
// ---------------------------------------------------------------------------
__global__ __launch_bounds__(256) void gatescell_kernel(
    const bf16* __restrict__ xbf, const bf16* __restrict__ hin,
    const bf16* __restrict__ wih, const bf16* __restrict__ whh,
    const float* __restrict__ bih, const float* __restrict__ bhh,
    float* __restrict__ c, bf16* __restrict__ hout) {
  __shared__ float sg[4][BATCH][32];                 // 32 KB gate tile

  const int lane = threadIdx.x & 31;
  const int wave = threadIdx.x >> 5;
  const int g    = wave >> 1;                        // gate 0..3 (i,f,g,o)
  const int j0   = blockIdx.x * 32;                  // hidden-col base
  const int joff = (wave & 1) * 16;
  const int n0   = g * HID + j0 + joff;              // row into W_ih/W_hh
  const int l = lane & 15, hi = lane >> 4;

  v8f acc[4] = {{}, {}, {}, {}};

  // prologue: first fragments of the x @ W_ih^T pass
  v16bf bcur = load_frag_b(wih, EMB, n0, 0, l, hi);
  v16bf a0 = load_frag_a(xbf, EMB,  0, 0, l, hi);
  v16bf a1 = load_frag_a(xbf, EMB, 16, 0, l, hi);
  v16bf a2 = load_frag_a(xbf, EMB, 32, 0, l, hi);
  v16bf a3 = load_frag_a(xbf, EMB, 48, 0, l, hi);

#pragma unroll
  for (int k0 = 0; k0 < EMB; k0 += 32) {             // x @ W_ih^T
    v16bf bn, f0, f1, f2, f3;
    if (k0 + 32 < EMB) {                             // prefetch next K-step
      bn = load_frag_b(wih, EMB, n0, k0 + 32, l, hi);
      f0 = load_frag_a(xbf, EMB,  0, k0 + 32, l, hi);
      f1 = load_frag_a(xbf, EMB, 16, k0 + 32, l, hi);
      f2 = load_frag_a(xbf, EMB, 32, k0 + 32, l, hi);
      f3 = load_frag_a(xbf, EMB, 48, k0 + 32, l, hi);
    } else {                                         // cross-loop prefetch
      bn = load_frag_b(whh, HID, n0, 0, l, hi);
      f0 = load_frag_a(hin, HID,  0, 0, l, hi);
      f1 = load_frag_a(hin, HID, 16, 0, l, hi);
      f2 = load_frag_a(hin, HID, 32, 0, l, hi);
      f3 = load_frag_a(hin, HID, 48, 0, l, hi);
    }
    acc[0] = WMMA_BF16(a0, bcur, acc[0]);
    acc[1] = WMMA_BF16(a1, bcur, acc[1]);
    acc[2] = WMMA_BF16(a2, bcur, acc[2]);
    acc[3] = WMMA_BF16(a3, bcur, acc[3]);
    bcur = bn; a0 = f0; a1 = f1; a2 = f2; a3 = f3;
  }
#pragma unroll
  for (int k0 = 0; k0 < HID; k0 += 32) {             // h @ W_hh^T
    v16bf bn = bcur, f0 = a0, f1 = a1, f2 = a2, f3 = a3;
    if (k0 + 32 < HID) {
      bn = load_frag_b(whh, HID, n0, k0 + 32, l, hi);
      f0 = load_frag_a(hin, HID,  0, k0 + 32, l, hi);
      f1 = load_frag_a(hin, HID, 16, k0 + 32, l, hi);
      f2 = load_frag_a(hin, HID, 32, k0 + 32, l, hi);
      f3 = load_frag_a(hin, HID, 48, k0 + 32, l, hi);
    }
    acc[0] = WMMA_BF16(a0, bcur, acc[0]);
    acc[1] = WMMA_BF16(a1, bcur, acc[1]);
    acc[2] = WMMA_BF16(a2, bcur, acc[2]);
    acc[3] = WMMA_BF16(a3, bcur, acc[3]);
    bcur = bn; a0 = f0; a1 = f1; a2 = f2; a3 = f3;
  }

  // stage biased gate values into LDS
  const int n = n0 + l;
  const float bias = bih[n] + bhh[n];
#pragma unroll
  for (int mt = 0; mt < 4; ++mt)
#pragma unroll
    for (int r = 0; r < 8; ++r)
      sg[g][mt * 16 + hi * 8 + r][joff + l] = acc[mt][r] + bias;

  __syncthreads();

  // in-block LSTM cell update over 64 rows x 32 cols
  for (int e = threadIdx.x; e < BATCH * 32; e += 256) {
    const int m = e >> 5, jj = e & 31;
    const int ci = m * HID + j0 + jj;
    const float ig = sg[0][m][jj], fg = sg[1][m][jj];
    const float gg = sg[2][m][jj], og = sg[3][m][jj];
    const float si = 1.f / (1.f + __expf(-ig));
    const float sf = 1.f / (1.f + __expf(-fg));
    const float so = 1.f / (1.f + __expf(-og));
    const float cn = sf * c[ci] + si * tanhf(gg);
    c[ci]    = cn;
    hout[ci] = (bf16)(so * tanhf(cn));
  }
}

// ---------------------------------------------------------------------------
// logits = h @ W_out^T + b_out -> d_out[:, t, :] + per-block argmax partials.
// 250 blocks x 256 thr; h staged in LDS (64 KB, reused for the reduction).
// ---------------------------------------------------------------------------
__global__ __launch_bounds__(256) void logits_kernel(
    const bf16* __restrict__ hbf, const bf16* __restrict__ wout,
    const float* __restrict__ bout, float* __restrict__ out, int t,
    float* __restrict__ pmax, int* __restrict__ pidx) {
  __shared__ union SU {
    bf16 a[BATCH * HID];                             // 64 KB h staging
    struct { float m[8 * BATCH]; int i[8 * BATCH]; } red;
  } su;

  {
    const uint4* src = (const uint4*)hbf;
    uint4* dst = (uint4*)su.a;
    const int nvec = BATCH * HID * (int)sizeof(bf16) / 16;   // 4096
    for (int i = threadIdx.x; i < nvec; i += 256) dst[i] = src[i];
  }
  __syncthreads();

  const int lane = threadIdx.x & 31;
  const int wave = threadIdx.x >> 5;
  const int n0   = blockIdx.x * 128 + wave * 16;
  const int l = lane & 15, hi = lane >> 4;

  auto lda = [&](int mtofs, int kk) -> v16bf {       // A-fragment from LDS
    const bf16* pa = su.a + (mtofs + l) * HID + kk + hi * 8;
    Frag f;
    f.h[0] = *(const v8bf*)(pa);
    f.h[1] = *(const v8bf*)(pa + 16);
    return f.v;
  };

  v8f acc[4] = {{}, {}, {}, {}};

  v16bf bcur = load_frag_b(wout, HID, n0, 0, l, hi);
  v16bf a0 = lda(0, 0), a1 = lda(16, 0), a2 = lda(32, 0), a3 = lda(48, 0);

#pragma unroll
  for (int k0 = 0; k0 < HID; k0 += 32) {
    v16bf bn = bcur, f0 = a0, f1 = a1, f2 = a2, f3 = a3;
    if (k0 + 32 < HID) {                             // prefetch next K-step
      bn = load_frag_b(wout, HID, n0, k0 + 32, l, hi);
      f0 = lda( 0, k0 + 32);
      f1 = lda(16, k0 + 32);
      f2 = lda(32, k0 + 32);
      f3 = lda(48, k0 + 32);
    }
    acc[0] = WMMA_BF16(a0, bcur, acc[0]);
    acc[1] = WMMA_BF16(a1, bcur, acc[1]);
    acc[2] = WMMA_BF16(a2, bcur, acc[2]);
    acc[3] = WMMA_BF16(a3, bcur, acc[3]);
    bcur = bn; a0 = f0; a1 = f1; a2 = f2; a3 = f3;
  }

  __syncthreads();                                   // done reading su.a

  const int n = n0 + l;
  const float bias = bout[n];
  const size_t rstride = TLEN * (size_t)VOCAB;
  const size_t tv = (size_t)t * VOCAB + n;

#pragma unroll
  for (int mt = 0; mt < 4; ++mt) {
#pragma unroll
    for (int r = 0; r < 8; ++r) {
      const int m = mt * 16 + hi * 8 + r;
      const float v = acc[mt][r] + bias;
      out[(size_t)m * rstride + tv] = v;
      // per-row argmax across this wave's 16 columns (16-lane half-wave)
      float bv = v;
      int   bi = n;
#pragma unroll
      for (int s = 1; s < 16; s <<= 1) {
        const float ov = __shfl_xor(bv, s, 32);
        const int   oi = __shfl_xor(bi, s, 32);
        if (ov > bv || (ov == bv && oi < bi)) { bv = ov; bi = oi; }
      }
      if (l == 0) {
        su.red.m[wave * BATCH + m] = bv;
        su.red.i[wave * BATCH + m] = bi;
      }
    }
  }
  __syncthreads();

  if (threadIdx.x < BATCH) {                         // cross-wave reduce
    const int m = threadIdx.x;
    float bv = -INFINITY;
    int   bi = 0x7fffffff;
#pragma unroll
    for (int w = 0; w < 8; ++w) {
      const float v = su.red.m[w * BATCH + m];
      const int   i = su.red.i[w * BATCH + m];
      if (v > bv || (v == bv && i < bi)) { bv = v; bi = i; }
    }
    pmax[(size_t)m * NLBLK + blockIdx.x] = bv;
    pidx[(size_t)m * NLBLK + blockIdx.x] = bi;
  }
}

// ---------------------------------------------------------------------------
// Reduce 250 per-block partials per batch row -> greedy token; gather emb.
// ---------------------------------------------------------------------------
__global__ __launch_bounds__(256) void argmax_finalize_kernel(
    const float* __restrict__ pmax, const int* __restrict__ pidx,
    const float* __restrict__ emb, bf16* __restrict__ xbf) {
  const int b = blockIdx.x;
  float best = -INFINITY;
  int bidx = 0x7fffffff;
  for (int p = threadIdx.x; p < NLBLK; p += 256) {
    const float v = pmax[(size_t)b * NLBLK + p];
    const int   i = pidx[(size_t)b * NLBLK + p];
    if (v > best || (v == best && i < bidx)) { best = v; bidx = i; }
  }
  __shared__ float smax[256];
  __shared__ int   sidx[256];
  smax[threadIdx.x] = best;
  sidx[threadIdx.x] = bidx;
  __syncthreads();
  for (int s = 128; s > 0; s >>= 1) {
    if (threadIdx.x < s) {
      const float o = smax[threadIdx.x + s];
      const int  oi = sidx[threadIdx.x + s];
      if (o > smax[threadIdx.x] ||
          (o == smax[threadIdx.x] && oi < sidx[threadIdx.x])) {
        smax[threadIdx.x] = o;
        sidx[threadIdx.x] = oi;
      }
    }
    __syncthreads();
  }
  const int idx = sidx[0];
  const float* erow = emb + (size_t)idx * EMB;
  for (int e = threadIdx.x; e < EMB; e += 256)
    xbf[(size_t)b * EMB + e] = (bf16)erow[e];
}

// ---------------------------------------------------------------------------
// Setup kernels
// ---------------------------------------------------------------------------
__global__ void cvt_bf16_kernel(const float* __restrict__ in,
                                bf16* __restrict__ out, int n) {
  for (int i = blockIdx.x * blockDim.x + threadIdx.x; i < n;
       i += gridDim.x * blockDim.x)
    out[i] = (bf16)in[i];
}

__global__ void init_state_kernel(const float* __restrict__ last_h,
                                  const float* __restrict__ bos,
                                  float* __restrict__ c,
                                  bf16* __restrict__ hbf,
                                  bf16* __restrict__ xbf) {
  const int i = blockIdx.x * blockDim.x + threadIdx.x;
  if (i >= BATCH * HID) return;
  const int e = i & (EMB - 1);
  c[i]   = 0.f;
  hbf[i] = (bf16)last_h[i];
  xbf[i] = (bf16)bos[e];                             // x0 = bos broadcast
}

__global__ void zero_kernel(float* __restrict__ p, int n) {
  const int i = blockIdx.x * blockDim.x + threadIdx.x;
  if (i < n) p[i] = 0.f;
}

// ---------------------------------------------------------------------------
extern "C" void kernel_launch(void* const* d_in, const int* in_sizes, int n_in,
                              void* d_out, int out_size, void* d_ws, size_t ws_size,
                              hipStream_t stream) {
  (void)in_sizes; (void)n_in; (void)out_size; (void)ws_size;
  const float* last_enc_h = (const float*)d_in[1];
  const float* bos        = (const float*)d_in[2];
  const float* emb        = (const float*)d_in[3];
  const float* W_ih       = (const float*)d_in[4];
  const float* W_hh       = (const float*)d_in[5];
  const float* b_ih       = (const float*)d_in[6];
  const float* b_hh       = (const float*)d_in[7];
  const float* W_out      = (const float*)d_in[8];
  const float* b_out      = (const float*)d_in[9];
  float* out = (float*)d_out;

  char* ws = (char*)d_ws;
  size_t off = 0;
  auto carve = [&](size_t bytes) {
    char* p = ws + off;
    off += (bytes + 255) & ~(size_t)255;
    return p;
  };
  bf16*  w_ih_b  = (bf16*)carve((size_t)G4 * EMB * sizeof(bf16));      // 2 MB
  bf16*  w_hh_b  = (bf16*)carve((size_t)G4 * HID * sizeof(bf16));      // 2 MB
  bf16*  w_out_b = (bf16*)carve((size_t)VOCAB * HID * sizeof(bf16));   // 32.8 MB
  float* c_st    = (float*)carve((size_t)BATCH * HID * sizeof(float));
  bf16*  hA      = (bf16*)carve((size_t)BATCH * HID * sizeof(bf16));
  bf16*  hB      = (bf16*)carve((size_t)BATCH * HID * sizeof(bf16));
  bf16*  xbf     = (bf16*)carve((size_t)BATCH * EMB * sizeof(bf16));
  float* pmax    = (float*)carve((size_t)BATCH * NLBLK * sizeof(float));
  int*   pidx    = (int*)carve((size_t)BATCH * NLBLK * sizeof(int));

  // one-time weight conversion (bf16 weights stay L2-resident afterwards)
  {
    const int n1 = G4 * EMB, n3 = VOCAB * HID;
    cvt_bf16_kernel<<<(n1 + 255) / 256, 256, 0, stream>>>(W_ih,  w_ih_b,  n1);
    cvt_bf16_kernel<<<(n1 + 255) / 256, 256, 0, stream>>>(W_hh,  w_hh_b,  n1);
    cvt_bf16_kernel<<<4096, 256, 0, stream>>>(W_out, w_out_b, n3);
  }
  init_state_kernel<<<(BATCH * HID + 255) / 256, 256, 0, stream>>>(
      last_enc_h, bos, c_st, hA, xbf);
  zero_kernel<<<(2 * BATCH * TLEN + 255) / 256, 256, 0, stream>>>(
      out + (size_t)BATCH * TLEN * VOCAB, 2 * BATCH * TLEN);

  // sequential greedy decode: 3 kernels per step
  for (int t = 0; t < TLEN; ++t) {
    const bf16* hin  = (t & 1) ? hB : hA;
    bf16*       hout = (t & 1) ? hA : hB;
    gatescell_kernel<<<HID / 32, 256, 0, stream>>>(xbf, hin, w_ih_b, w_hh_b,
                                                   b_ih, b_hh, c_st, hout);
    logits_kernel<<<NLBLK, 256, 0, stream>>>(hout, w_out_b, b_out, out, t,
                                             pmax, pidx);
    argmax_finalize_kernel<<<BATCH, 256, 0, stream>>>(pmax, pidx, emb, xbf);
  }
}
